// PagedAttentionBlock_90580860272708
// MI455X (gfx1250) — compile-verified, matrix operand order checked by reading, not run on hardware
//
#include <hip/hip_runtime.h>

// ---------------- problem constants (from reference) ----------------
static constexpr int kB    = 4;
static constexpr int kLin  = 512;
static constexpr int kH    = 32;
static constexpr int kD    = 128;
static constexpr int kBlk  = 16;
static constexpr int kCtx  = 1024;           // L_IN + L_C
static constexpr int kNB   = kB * kCtx / kBlk; // 256 blocks
static constexpr int kT    = kB * kLin;        // 2048 tokens

// ---------------- types ----------------
typedef __attribute__((ext_vector_type(16))) __bf16 bf16x16;
typedef __attribute__((ext_vector_type(8)))  float  f32x8;

struct alignas(16) U4 { unsigned int x, y, z, w; };
struct alignas(16) F4 { float x, y, z, w; };
union H8   { unsigned short h[8]; U4 u; };
union Frag { bf16x16 b; U4 u[2]; };

__device__ __forceinline__ unsigned short f2bf(float f) {
  unsigned int u = __float_as_uint(f);
  u = (u + 0x7FFFu + ((u >> 16) & 1u)) >> 16;   // round-to-nearest-even
  return (unsigned short)u;
}

// Fragment-major layouts (bf16, 2048 halfs per (block,h) / (b,h,qt)):
//  K/Q  : [..][c(4)][seg(2)][lane(32)][j(8)]   value = X[m = lane&15][d = c*32 + seg*16 + 8*(lane>>4) + j]
//  V^T  : [..][dt(8)][lane(32)][j(8)]          value = V[key = 8*(lane>>4) + j][d = dt*16 + (lane&15)]

// ---------------- kernel 1: convert caches to fragment layout ----------------
__global__ __launch_bounds__(256) void cache_to_frag_kernel(
    const float* __restrict__ Kc, const float* __restrict__ Vc,
    unsigned short* __restrict__ Kf, unsigned short* __restrict__ Vf) {
  int idx = blockIdx.x * 256 + threadIdx.x;          // (blk, off, h)
  int h   = idx & (kH - 1);
  int off = (idx >> 5) & (kBlk - 1);
  int blk = idx >> 9;
  const float* krow = Kc + (((size_t)(blk * kBlk + off)) * kH + h) * kD;
  const float* vrow = Vc + (((size_t)(blk * kBlk + off)) * kH + h) * kD;

  unsigned short* kb = Kf + ((size_t)blk * kH + h) * 2048;
#pragma unroll
  for (int run = 0; run < 16; ++run) {               // run = c*4 + seg*2 + lb ; d0 = run*8
    int c = run >> 2, seg = (run >> 1) & 1, lb = run & 1;
    int lane = off + 16 * lb;
    H8 t;
#pragma unroll
    for (int j = 0; j < 8; ++j) t.h[j] = f2bf(krow[run * 8 + j]);
    *reinterpret_cast<U4*>(kb + ((c * 2 + seg) * 256 + lane * 8)) = t.u;
  }

  unsigned short* vb = Vf + ((size_t)blk * kH + h) * 2048;
  int lb2 = (off >> 3) & 1, j2 = off & 7;
#pragma unroll 4
  for (int d = 0; d < kD; ++d) {
    vb[(((d >> 4) * 32) + (d & 15) + 16 * lb2) * 8 + j2] = f2bf(vrow[d]);
  }
}

// ---------------- kernel 2: RoPE + scatter new K/V + build scaled Q fragments ----------------
__global__ __launch_bounds__(256) void rope_scatter_kernel(
    const float* __restrict__ Q, const float* __restrict__ K, const float* __restrict__ V,
    const float* __restrict__ Cs, const float* __restrict__ Sn,
    const int* __restrict__ slots,
    unsigned short* __restrict__ Qf, unsigned short* __restrict__ Kf,
    unsigned short* __restrict__ Vf) {
  int idx = blockIdx.x * 256 + threadIdx.x;          // t*H + h
  int h = idx & (kH - 1);
  int t = idx >> 5;
  const float* q  = Q  + ((size_t)t * kH + h) * kD;
  const float* k  = K  + ((size_t)t * kH + h) * kD;
  const float* v  = V  + ((size_t)t * kH + h) * kD;
  const float* cs = Cs + (size_t)t * kD;
  const float* sn = Sn + (size_t)t * kD;

  int b = t / kLin, qi = t % kLin;
  int qt = qi >> 4, m = qi & 15;
  unsigned short* qb = Qf + (((size_t)(b * kH + h)) * 32 + qt) * 2048;

  int slot = slots[t];
  int sblk = slot >> 4, soff = slot & 15;
  unsigned short* kb = Kf + ((size_t)sblk * kH + h) * 2048;

  const float scale = 0.08838834764831845f;          // 1/sqrt(128)
#pragma unroll
  for (int run = 0; run < 16; ++run) {
    int c = run >> 2, seg = (run >> 1) & 1, lb = run & 1;
    float sign = (run < 8) ? -1.0f : 1.0f;           // rot_half sign
    H8 tq, tk;
#pragma unroll
    for (int j = 0; j < 8; ++j) {
      int d = run * 8 + j;
      int dp = d ^ 64;
      float qr = q[d] * cs[d] + sign * q[dp] * sn[d];
      float kr = k[d] * cs[d] + sign * k[dp] * sn[d];
      tq.h[j] = f2bf(qr * scale);
      tk.h[j] = f2bf(kr);
    }
    *reinterpret_cast<U4*>(qb + ((c * 2 + seg) * 256 + (m    + 16 * lb) * 8)) = tq.u;
    *reinterpret_cast<U4*>(kb + ((c * 2 + seg) * 256 + (soff + 16 * lb) * 8)) = tk.u;
  }

  unsigned short* vb = Vf + ((size_t)sblk * kH + h) * 2048;
  int lb2 = (soff >> 3) & 1, j2 = soff & 7;
#pragma unroll 4
  for (int d = 0; d < kD; ++d) {
    vb[(((d >> 4) * 32) + (d & 15) + 16 * lb2) * 8 + j2] = f2bf(v[d]);
  }
}

// ---------------- kernel 3: flash attention (S^T / O^T formulation) ----------------
__global__ __launch_bounds__(256) void paged_attn_kernel(
    const unsigned short* __restrict__ Qf, const unsigned short* __restrict__ Kf,
    const unsigned short* __restrict__ Vf, const float* __restrict__ maskp,
    const int* __restrict__ input_length, const int* __restrict__ cache_length,
    const int* __restrict__ block_tables, float* __restrict__ out) {
  const int lane = threadIdx.x & 31;
  const int wv   = threadIdx.x >> 5;                 // wave id in WG (0..7)
  int qc = blockIdx.x & 3;                           // q-chunk of 128 rows
  int rest = blockIdx.x >> 2;
  int h = rest & (kH - 1);
  int b = rest >> 5;
  int qt = qc * 8 + wv;                              // 16-row q tile (0..31)
  const int lh = lane >> 4;                          // lane half
  const int qm = lane & 15;                          // q row within tile (per lane)

  // Q^T B-fragments, resident across the whole k loop (scale pre-folded)
  Frag qf[4];
  const unsigned short* qb = Qf + (((size_t)(b * kH + h)) * 32 + qt) * 2048;
#pragma unroll
  for (int c = 0; c < 4; ++c) {
    qf[c].u[0] = *reinterpret_cast<const U4*>(qb + (c * 2 + 0) * 256 + lane * 8);
    qf[c].u[1] = *reinterpret_cast<const U4*>(qb + (c * 2 + 1) * 256 + lane * 8);
  }

  int Lc  = cache_length[b];
  int ctx = Lc + input_length[b];
  int qglob = qt * 16 + qm;
  int limit = Lc + qglob;            // last valid kpos for this lane's q row
  if (limit > ctx - 1) limit = ctx - 1;
  int kmax = Lc + qt * 16 + 15;      // last valid kpos for the whole tile
  if (kmax > ctx - 1) kmax = ctx - 1;
  int nch = (kmax + 32) >> 5;        // 32-key chunks to process
  const int* bt = block_tables + b * (kCtx / kBlk);
  const float* mrow = maskp + (size_t)qglob * kCtx;

  f32x8 zero = {};
  f32x8 acc[8];
#pragma unroll
  for (int i = 0; i < 8; ++i) acc[i] = zero;
  float mrun = -3.0e38f, lrun = 0.0f;

  for (int kc2 = 0; kc2 < nch; ++kc2) {
    int kbase = kc2 << 5;
    int bl0 = bt[kc2 * 2 + 0];
    int bl1 = bt[kc2 * 2 + 1];
    const unsigned short* kb0 = Kf + ((size_t)bl0 * kH + h) * 2048;
    const unsigned short* kb1 = Kf + ((size_t)bl1 * kH + h) * 2048;

    // S^T tiles: A = K rows (M=key), B = Q^T (N=q); accumulate over D in 4 chunks of 32
    f32x8 s0 = zero, s1 = zero;
#pragma unroll
    for (int c = 0; c < 4; ++c) {
      Frag a0, a1;
      a0.u[0] = *reinterpret_cast<const U4*>(kb0 + (c * 2 + 0) * 256 + lane * 8);
      a0.u[1] = *reinterpret_cast<const U4*>(kb0 + (c * 2 + 1) * 256 + lane * 8);
      a1.u[0] = *reinterpret_cast<const U4*>(kb1 + (c * 2 + 0) * 256 + lane * 8);
      a1.u[1] = *reinterpret_cast<const U4*>(kb1 + (c * 2 + 1) * 256 + lane * 8);
      s0 = __builtin_amdgcn_wmma_f32_16x16x32_bf16(false, a0.b, false, qf[c].b, (short)0, s0, false, false);
      s1 = __builtin_amdgcn_wmma_f32_16x16x32_bf16(false, a1.b, false, qf[c].b, (short)0, s1, false, false);
    }

    // masking + online softmax (lane L pairs with lane L^16; both hold same q row)
    int k0 = kbase + lh * 8;
    float p0[8], p1[8];
    float tmax = -3.0e38f;
#pragma unroll
    for (int r = 0; r < 8; ++r) {
      int key0 = k0 + r, key1 = k0 + 16 + r;
      float v0 = s0[r] + mrow[key0];
      float v1 = s1[r] + mrow[key1];
      v0 = (key0 <= limit) ? v0 : -3.0e38f;
      v1 = (key1 <= limit) ? v1 : -3.0e38f;
      p0[r] = v0; p1[r] = v1;
      tmax = fmaxf(tmax, fmaxf(v0, v1));
    }
    tmax = fmaxf(tmax, __shfl_xor(tmax, 16, 32));
    float mnew  = fmaxf(mrun, tmax);
    float alpha = __expf(mrun - mnew);
    float psum = 0.0f;
    Frag pf;                                   // P^T B-fragment: built in-place, no shuffles
#pragma unroll
    for (int r = 0; r < 8; ++r) {
      float e0 = __expf(p0[r] - mnew);
      float e1 = __expf(p1[r] - mnew);
      psum += e0 + e1;
      pf.b[r]     = (__bf16)e0;
      pf.b[r + 8] = (__bf16)e1;
    }
    psum += __shfl_xor(psum, 16, 32);
    lrun = lrun * alpha + psum;
    mrun = mnew;

    // O^T += V^T (A, M=d) x P^T (B, N=q)
    const unsigned short* vb0 = Vf + ((size_t)bl0 * kH + h) * 2048;
    const unsigned short* vb1 = Vf + ((size_t)bl1 * kH + h) * 2048;
#pragma unroll
    for (int dt = 0; dt < 8; ++dt) {
      Frag vf2;
      vf2.u[0] = *reinterpret_cast<const U4*>(vb0 + dt * 256 + lane * 8);
      vf2.u[1] = *reinterpret_cast<const U4*>(vb1 + dt * 256 + lane * 8);
      acc[dt] = acc[dt] * alpha;
      acc[dt] = __builtin_amdgcn_wmma_f32_16x16x32_bf16(false, vf2.b, false, pf.b, (short)0, acc[dt], false, false);
    }
  }

  // epilogue: normalize and store O (lane = q row, reg = d)
  float rl = 1.0f / lrun;
  float* orow = out + (((size_t)(b * kLin + qglob)) * kH + h) * kD + lh * 8;
#pragma unroll
  for (int dt = 0; dt < 8; ++dt) {
    F4 o0 = { acc[dt][0] * rl, acc[dt][1] * rl, acc[dt][2] * rl, acc[dt][3] * rl };
    F4 o1 = { acc[dt][4] * rl, acc[dt][5] * rl, acc[dt][6] * rl, acc[dt][7] * rl };
    *reinterpret_cast<F4*>(orow + dt * 16)     = o0;
    *reinterpret_cast<F4*>(orow + dt * 16 + 4) = o1;
  }
}

// ---------------- launch ----------------
extern "C" void kernel_launch(void* const* d_in, const int* in_sizes, int n_in,
                              void* d_out, int out_size, void* d_ws, size_t ws_size,
                              hipStream_t stream) {
  const float* Q    = (const float*)d_in[0];
  const float* K    = (const float*)d_in[1];
  const float* V    = (const float*)d_in[2];
  const float* Kc   = (const float*)d_in[3];
  const float* Vc   = (const float*)d_in[4];
  const float* Cs   = (const float*)d_in[5];
  const float* Sn   = (const float*)d_in[6];
  const float* mask = (const float*)d_in[7];
  const int* in_len = (const int*)d_in[8];
  const int* c_len  = (const int*)d_in[9];
  const int* slots  = (const int*)d_in[10];
  const int* btab   = (const int*)d_in[11];

  // scratch: Qf (16MB) | Kf (32MB) | Vf (32MB), bf16 stored as ushort
  unsigned short* Qf = (unsigned short*)d_ws;
  unsigned short* Kf = Qf + (size_t)kT * kH * kD;          // 8,388,608 halfs
  unsigned short* Vf = Kf + (size_t)kNB * kBlk * kH * kD;  // +16,777,216 halfs

  cache_to_frag_kernel<<<(kNB * kBlk * kH) / 256, 256, 0, stream>>>(Kc, Vc, Kf, Vf);
  rope_scatter_kernel<<<(kT * kH) / 256, 256, 0, stream>>>(Q, K, V, Cs, Sn, slots, Qf, Kf, Vf);
  paged_attn_kernel<<<kB * kH * (kLin / 128), 256, 0, stream>>>(Qf, Kf, Vf, mask, in_len, c_len, btab,
                                                                (float*)d_out);
}
